// DiffiT_66468913873456
// MI455X (gfx1250) — matrix operand order repaired and checked by reading, not compiled, not run on hardware
//
#include <hip/hip_runtime.h>
#include <math.h>

// ---------------------------------------------------------------- types
typedef _Float16 half_t;
typedef _Float16 v16h __attribute__((ext_vector_type(16)));
typedef _Float16 v8h  __attribute__((ext_vector_type(8)));
typedef float    v8f  __attribute__((ext_vector_type(8)));

union V16 { v16h v; v8h h2[2]; };

// ---------------------------------------------------------------- problem dims
#define TOK_N   256           // tokens per window
#define CDIM    1152
#define HEADS   16
#define DH      72
#define DHP     96            // DH padded to 3*32 for WMMA K-loop
#define MLPDIM  4608
#define BATCH   64
#define ROWS    (BATCH * TOK_N)   // 16384

// GEMM tile config: 128x128 block tile, K-step 64 (2 WMMA slices), double-buffered LDS
#define BM 128
#define BN 128
#define BK2 64
#define LDP2 72               // padded LDS row stride in halves (144 B, 16-B aligned)

enum { EPI_TADD = 0, EPI_QKV = 1, EPI_PROJ = 2, EPI_FC1 = 3, EPI_FC2 = 4 };

// ---------------------------------------------------------------- helpers
__device__ __forceinline__ v8f wmma_f16(v16h a, v16h b, v8f c) {
    return __builtin_amdgcn_wmma_f32_16x16x32_f16(false, a, false, b, (short)0, c,
                                                  false, false);
}

__device__ __forceinline__ float gelu_tanh(float x) {
    return 0.5f * x * (1.0f + tanhf(0.7978845608028654f * (x + 0.044715f * x * x * x)));
}

// ---------------------------------------------------------------- small kernels
__global__ __launch_bounds__(256) void cvt_f32_f16(const float* __restrict__ s,
                                                   half_t* __restrict__ d, size_t n) {
    size_t i = (size_t)blockIdx.x * 256 + threadIdx.x;
    size_t st = (size_t)gridDim.x * 256;
    for (; i < n; i += st) d[i] = (half_t)s[i];
}

__global__ __launch_bounds__(256) void fill_zero_u32(unsigned int* __restrict__ d, size_t n) {
    size_t i = (size_t)blockIdx.x * 256 + threadIdx.x;
    size_t st = (size_t)gridDim.x * 256;
    for (; i < n; i += st) d[i] = 0u;
}

// pad c [64,1152] -> [128,1152] f16 (rows 64..127 = 0)
__global__ __launch_bounds__(256) void cpad_kernel(const float* __restrict__ c,
                                                   half_t* __restrict__ out) {
    int i = blockIdx.x * 256 + threadIdx.x;       // 128*1152 threads
    int row = i / CDIM;
    int col = i - row * CDIM;
    out[i] = (row < BATCH) ? (half_t)c[row * CDIM + col] : (half_t)0.0f;
}

// biasmat[h][i][j] = rpb_table[rel_idx[i,j]*H + h]   (shared across all batches)
__global__ __launch_bounds__(256) void bias_pre_kernel(const float* __restrict__ rpb,
                                                       const int* __restrict__ rel,
                                                       float* __restrict__ bm) {
    size_t t = (size_t)blockIdx.x * 256 + threadIdx.x;
    if (t >= (size_t)HEADS * TOK_N * TOK_N) return;
    int j = (int)(t & 255);
    int i = (int)((t >> 8) & 255);
    int h = (int)(t >> 16);
    bm[t] = rpb[rel[i * TOK_N + j] * HEADS + h];
}

// LayerNorm over last dim (C=1152), f32 in -> f16 out
__global__ __launch_bounds__(256) void ln_kernel(const float* __restrict__ x,
                                                 half_t* __restrict__ out) {
    const int row = blockIdx.x;
    const int tid = threadIdx.x;
    const float* xr = x + (size_t)row * CDIM;
    float s = 0.f, s2 = 0.f;
    for (int i = tid; i < CDIM; i += 256) { float v = xr[i]; s += v; s2 += v * v; }
    for (int m = 1; m < 32; m <<= 1) { s += __shfl_xor(s, m, 32); s2 += __shfl_xor(s2, m, 32); }
    __shared__ float rs[8], rs2[8];
    int wave = tid >> 5, lane = tid & 31;
    if (lane == 0) { rs[wave] = s; rs2[wave] = s2; }
    __syncthreads();
    if (wave == 0) {
        float a = (lane < 8) ? rs[lane] : 0.f;
        float b = (lane < 8) ? rs2[lane] : 0.f;
        for (int m = 1; m < 8; m <<= 1) { a += __shfl_xor(a, m, 32); b += __shfl_xor(b, m, 32); }
        if (lane == 0) { rs[0] = a; rs2[0] = b; }
    }
    __syncthreads();
    const float mean = rs[0] * (1.0f / CDIM);
    const float var  = rs2[0] * (1.0f / CDIM) - mean * mean;
    const float inv  = rsqrtf(var + 1e-6f);
    half_t* orow = out + (size_t)row * CDIM;
    for (int i = tid; i < CDIM; i += 256) orow[i] = (half_t)((xr[i] - mean) * inv);
}

// ---------------------------------------------------------------- generic WMMA GEMM
// out[m,n] = sum_k A[m,k] * W[n,k]   (A f16 [M,K] row-major, W f16 [N,K] row-major)
// 256 threads = 8 waves; block tile 128x128; K-step 64; wave tile 64x32 = 4x2 WMMA tiles.
// Double-buffered LDS: global->VGPR for step k+1 issued before the WMMA clause of step k,
// VGPR->LDS after it; one barrier per K-step. Next-next tile prefetched into L2.
template <int EPI>
__global__ __launch_bounds__(256) void gemm_f16_wmma(
    const half_t* __restrict__ A, const half_t* __restrict__ W,
    const float* __restrict__ bias, const float* __restrict__ addv,
    const float* __restrict__ resid, float* __restrict__ outf,
    half_t* __restrict__ outh,
    half_t* __restrict__ qo, half_t* __restrict__ ko, half_t* __restrict__ vto,
    int M, int N, int K)
{
    __shared__ __align__(16) half_t As[2][BM][LDP2];
    __shared__ __align__(16) half_t Bs[2][BN][LDP2];

    const int tid  = threadIdx.x;
    const int wave = tid >> 5, lane = tid & 31;
    const int l15  = lane & 15;
    const int hi   = lane >> 4;            // 0: lanes 0-15, 1: lanes 16-31
    const int kbA  = hi * 8;               // A-frag K base (interleaved layout)
    const int kbB  = hi * 8;               // B-frag K base (interleaved layout)
    const int wm   = wave >> 2;            // 0..1
    const int wn   = wave & 3;             // 0..3
    const int m0   = blockIdx.y * BM;
    const int n0   = blockIdx.x * BN;

    // staging map: 128 rows x 64 halves per matrix = 1024 x v8h chunks / 256 threads
    const int srow = tid >> 3;             // rows for chunk i: srow + 32*i? no: chunk=tid+256*i
    (void)srow;

    v8f acc[4][2];
    const v8f vzero = {0.f, 0.f, 0.f, 0.f, 0.f, 0.f, 0.f, 0.f};
#pragma unroll
    for (int mt = 0; mt < 4; ++mt)
#pragma unroll
        for (int nt = 0; nt < 2; ++nt) acc[mt][nt] = vzero;

    v8h ra[4], rb[4];
    // ---- preload K-step 0 ----
#pragma unroll
    for (int i = 0; i < 4; ++i) {
        int chunk = tid + 256 * i;
        int row = chunk >> 3;
        int kc  = (chunk & 7) * 8;
        ra[i] = *(const v8h*)(A + (size_t)(m0 + row) * K + kc);
        rb[i] = *(const v8h*)(W + (size_t)(n0 + row) * K + kc);
    }
#pragma unroll
    for (int i = 0; i < 4; ++i) {
        int chunk = tid + 256 * i;
        int row = chunk >> 3;
        int kc  = (chunk & 7) * 8;
        *(v8h*)&As[0][row][kc] = ra[i];
        *(v8h*)&Bs[0][row][kc] = rb[i];
    }

    int buf = 0;
    for (int k0 = 0; k0 < K; k0 += BK2) {
        __syncthreads();
        const bool more = (k0 + BK2) < K;
        if (more) {
            // issue next tile's global loads; latency hides under the WMMA clause
#pragma unroll
            for (int i = 0; i < 4; ++i) {
                int chunk = tid + 256 * i;
                int row = chunk >> 3;
                int kc  = (chunk & 7) * 8;
                ra[i] = *(const v8h*)(A + (size_t)(m0 + row) * K + (k0 + BK2) + kc);
                rb[i] = *(const v8h*)(W + (size_t)(n0 + row) * K + (k0 + BK2) + kc);
            }
            if (k0 + 2 * BK2 < K) {
                // pull tile-after-next toward L2 (global_prefetch_b8)
                int row = tid >> 3;
                __builtin_prefetch(A + (size_t)(m0 + row) * K + (k0 + 2 * BK2), 0, 1);
                __builtin_prefetch(W + (size_t)(n0 + row) * K + (k0 + 2 * BK2), 0, 1);
            }
        }

        // ---- 2 WMMA K-slices out of the staged 64-deep tile ----
#pragma unroll
        for (int kk = 0; kk < 2; ++kk) {
            const int ko = kk * 32;
            V16 af[4], bf[2];
#pragma unroll
            for (int mt = 0; mt < 4; ++mt) {
                int r = wm * 64 + mt * 16 + l15;
                af[mt].h2[0] = *(const v8h*)&As[buf][r][ko + kbA];
                af[mt].h2[1] = *(const v8h*)&As[buf][r][ko + kbA + 16];
            }
#pragma unroll
            for (int nt = 0; nt < 2; ++nt) {
                int cn = wn * 32 + nt * 16 + l15;
                bf[nt].h2[0] = *(const v8h*)&Bs[buf][cn][ko + kbB];
                bf[nt].h2[1] = *(const v8h*)&Bs[buf][cn][ko + kbB + 16];
            }
#pragma unroll
            for (int mt = 0; mt < 4; ++mt)
#pragma unroll
                for (int nt = 0; nt < 2; ++nt)
                    acc[mt][nt] = wmma_f16(af[mt].v, bf[nt].v, acc[mt][nt]);
        }

        if (more) {
#pragma unroll
            for (int i = 0; i < 4; ++i) {
                int chunk = tid + 256 * i;
                int row = chunk >> 3;
                int kc  = (chunk & 7) * 8;
                *(v8h*)&As[buf ^ 1][row][kc] = ra[i];
                *(v8h*)&Bs[buf ^ 1][row][kc] = rb[i];
            }
        }
        buf ^= 1;
    }

    // ---------------- epilogue ----------------
#pragma unroll
    for (int mt = 0; mt < 4; ++mt) {
#pragma unroll
        for (int nt = 0; nt < 2; ++nt) {
            const int gn = n0 + wn * 32 + nt * 16 + l15;
            const float bn = bias[gn];
            int jj = 0, hh = 0, dd = 0;
            if (EPI == EPI_QKV) {
                jj = gn / CDIM;
                int rem = gn - jj * CDIM;
                hh = rem / DH;
                dd = rem - hh * DH;
            }
#pragma unroll
            for (int r = 0; r < 8; ++r) {
                const int gm = m0 + wm * 64 + mt * 16 + hi * 8 + r;
                float v = acc[mt][nt][r] + bn;
                if (EPI == EPI_TADD) {
                    outf[(size_t)gm * N + gn] = v;
                } else if (EPI == EPI_QKV) {
                    const int b = gm >> 8, tok = gm & 255;
                    v += addv[(size_t)b * N + gn];
                    const int bh = b * HEADS + hh;
                    if (jj == 0)
                        qo[((size_t)bh * TOK_N + tok) * DHP + dd] = (half_t)v;
                    else if (jj == 1)
                        ko[((size_t)bh * TOK_N + tok) * DHP + dd] = (half_t)v;
                    else
                        vto[((size_t)bh * DHP + dd) * TOK_N + tok] = (half_t)v;
                } else if (EPI == EPI_PROJ) {
                    outf[(size_t)gm * N + gn] = v + resid[(size_t)gm * N + gn];
                } else if (EPI == EPI_FC1) {
                    outh[(size_t)gm * N + gn] = (half_t)gelu_tanh(v);
                } else { // EPI_FC2
                    outf[(size_t)gm * N + gn] = v + resid[(size_t)gm * N + gn];
                }
            }
        }
    }
}

// ---------------------------------------------------------------- attention
// One block per (b,h): 128 threads = 4 waves, 4 passes; each wave/pass owns 16 queries.
// Stage 1: S[16x256] = q·k^T (K=96, 3 WMMA K-steps, 16 key tiles) + bias, softmax.
// Stage 2: O[16x96] = P·vT (K=256, 8 K-steps, 6 d-tiles), store d<72 as f16 into o.
__global__ __launch_bounds__(128) void attn_kernel(
    const half_t* __restrict__ q, const half_t* __restrict__ k,
    const half_t* __restrict__ vT, const float* __restrict__ biasm,
    half_t* __restrict__ o)
{
    __shared__ __align__(16) half_t Pb[4][16 * 264];   // per-wave P tile, padded rows

    const int bh = blockIdx.x;
    const int b  = bh >> 4, h = bh & 15;
    const int tid = threadIdx.x;
    const int wave = tid >> 5, lane = tid & 31;
    const int l15 = lane & 15;
    const int hi  = lane >> 4;
    const int kbA = hi * 8;     // A-frag K base
    const int kbB = hi * 8;     // B-frag K base

    const half_t* qp = q  + (size_t)bh * TOK_N * DHP;
    const half_t* kp = k  + (size_t)bh * TOK_N * DHP;
    const half_t* vp = vT + (size_t)bh * DHP * TOK_N;
    const float*  bp = biasm + (size_t)h * TOK_N * TOK_N;

    const float scale = 0.11785113019775793f;   // 1/sqrt(72)
    const v8f vzero = {0.f, 0.f, 0.f, 0.f, 0.f, 0.f, 0.f, 0.f};

    for (int pass = 0; pass < 4; ++pass) {
        const int qbase = pass * 64 + wave * 16;

        // ---- stage 1: scores ----
        v8f S[16];
#pragma unroll
        for (int t = 0; t < 16; ++t) S[t] = vzero;

#pragma unroll
        for (int ks = 0; ks < 3; ++ks) {
            V16 af;
            const half_t* qrow = qp + (size_t)(qbase + l15) * DHP + ks * 32;
            af.h2[0] = *(const v8h*)(qrow + kbA);
            af.h2[1] = *(const v8h*)(qrow + kbA + 16);
#pragma unroll
            for (int t = 0; t < 16; ++t) {
                V16 bf;
                const half_t* krow = kp + (size_t)(t * 16 + l15) * DHP + ks * 32;
                bf.h2[0] = *(const v8h*)(krow + kbB);
                bf.h2[1] = *(const v8h*)(krow + kbB + 16);
                S[t] = wmma_f16(af.v, bf.v, S[t]);
            }
        }

        // ---- bias + softmax (rows live in 16-lane groups) ----
#pragma unroll
        for (int r = 0; r < 8; ++r) {
            const int tok = qbase + hi * 8 + r;
            float mx = -1e30f;
#pragma unroll
            for (int t = 0; t < 16; ++t) {
                float s = S[t][r] * scale + bp[(size_t)tok * TOK_N + t * 16 + l15];
                S[t][r] = s;
                mx = fmaxf(mx, s);
            }
            for (int m = 1; m < 16; m <<= 1) mx = fmaxf(mx, __shfl_xor(mx, m, 32));
            float sum = 0.f;
#pragma unroll
            for (int t = 0; t < 16; ++t) {
                float e = __expf(S[t][r] - mx);
                S[t][r] = e;
                sum += e;
            }
            for (int m = 1; m < 16; m <<= 1) sum += __shfl_xor(sum, m, 32);
            const float inv = 1.0f / sum;
#pragma unroll
            for (int t = 0; t < 16; ++t)
                Pb[wave][(hi * 8 + r) * 264 + t * 16 + l15] = (half_t)(S[t][r] * inv);
        }
        __syncthreads();

        // ---- stage 2: O = P @ vT ----
        v8f O[6];
#pragma unroll
        for (int nt = 0; nt < 6; ++nt) O[nt] = vzero;

#pragma unroll
        for (int ks = 0; ks < 8; ++ks) {
            V16 af;
            const half_t* prow = &Pb[wave][l15 * 264 + ks * 32];
            af.h2[0] = *(const v8h*)(prow + kbA);
            af.h2[1] = *(const v8h*)(prow + kbA + 16);
#pragma unroll
            for (int nt = 0; nt < 6; ++nt) {
                V16 bf;
                const half_t* vrow = vp + (size_t)(nt * 16 + l15) * TOK_N + ks * 32;
                bf.h2[0] = *(const v8h*)(vrow + kbB);
                bf.h2[1] = *(const v8h*)(vrow + kbB + 16);
                O[nt] = wmma_f16(af.v, bf.v, O[nt]);
            }
        }

        // ---- store (only d < 72) ----
#pragma unroll
        for (int nt = 0; nt < 6; ++nt) {
            const int d = nt * 16 + l15;
            if (d < DH) {
#pragma unroll
                for (int r = 0; r < 8; ++r) {
                    const int tok = qbase + hi * 8 + r;
                    o[((size_t)(b * TOK_N + tok)) * CDIM + h * DH + d] = (half_t)O[nt][r];
                }
            }
        }
        __syncthreads();
    }
}

// ---------------------------------------------------------------- host launch
extern "C" void kernel_launch(void* const* d_in, const int* in_sizes, int n_in,
                              void* d_out, int out_size, void* d_ws, size_t ws_size,
                              hipStream_t stream) {
    (void)in_sizes; (void)n_in; (void)out_size; (void)ws_size;
    const float* x       = (const float*)d_in[0];
    const float* c       = (const float*)d_in[1];
    const float* qkv_w   = (const float*)d_in[2];
    const float* qkv_b   = (const float*)d_in[3];
    const float* qkvt_w  = (const float*)d_in[4];
    const float* qkvt_b  = (const float*)d_in[5];
    const float* rpb     = (const float*)d_in[6];
    const float* proj_w  = (const float*)d_in[7];
    const float* proj_b  = (const float*)d_in[8];
    const float* fc1_w   = (const float*)d_in[9];
    const float* fc1_b   = (const float*)d_in[10];
    const float* fc2_w   = (const float*)d_in[11];
    const float* fc2_b   = (const float*)d_in[12];
    const int*   rel_idx = (const int*)d_in[13];
    float* out = (float*)d_out;

    // workspace carve-up (256-byte aligned)
    char* base = (char*)d_ws;
    size_t off = 0;
    auto carve = [&](size_t bytes) -> char* {
        char* p = base + off;
        off = (off + bytes + 255) & ~(size_t)255;
        return p;
    };
    const size_t QSZ = (size_t)BATCH * HEADS * TOK_N * DHP * sizeof(half_t); // 50.3 MB
    half_t* wqkv  = (half_t*)carve((size_t)3 * CDIM * CDIM * sizeof(half_t));
    half_t* wqkvt = (half_t*)carve((size_t)3 * CDIM * CDIM * sizeof(half_t));
    half_t* wproj = (half_t*)carve((size_t)CDIM * CDIM * sizeof(half_t));
    half_t* wfc1  = (half_t*)carve((size_t)MLPDIM * CDIM * sizeof(half_t));
    half_t* wfc2  = (half_t*)carve((size_t)MLPDIM * CDIM * sizeof(half_t));
    half_t* cpad  = (half_t*)carve((size_t)128 * CDIM * sizeof(half_t));
    float*  tadd  = (float*)carve((size_t)128 * 3 * CDIM * sizeof(float));
    half_t* hbuf  = (half_t*)carve((size_t)ROWS * CDIM * sizeof(half_t));   // h / h2
    half_t* qbuf  = (half_t*)carve(QSZ);
    half_t* kbuf  = (half_t*)carve(QSZ);
    half_t* vtb   = (half_t*)carve(QSZ);
    half_t* obuf  = (half_t*)carve((size_t)ROWS * CDIM * sizeof(half_t));
    float*  x1    = (float*)carve((size_t)ROWS * CDIM * sizeof(float));
    float*  biasm = (float*)carve((size_t)HEADS * TOK_N * TOK_N * sizeof(float));
    half_t* m1    = qbuf;   // reuse q+k+vT region (exactly 3*QSZ = 151 MB) for MLP act

    // 1) weight conversion f32 -> f16
    cvt_f32_f16<<<4096, 256, 0, stream>>>(qkv_w,  wqkv,  (size_t)3 * CDIM * CDIM);
    cvt_f32_f16<<<4096, 256, 0, stream>>>(qkvt_w, wqkvt, (size_t)3 * CDIM * CDIM);
    cvt_f32_f16<<<2048, 256, 0, stream>>>(proj_w, wproj, (size_t)CDIM * CDIM);
    cvt_f32_f16<<<4096, 256, 0, stream>>>(fc1_w,  wfc1,  (size_t)MLPDIM * CDIM);
    cvt_f32_f16<<<4096, 256, 0, stream>>>(fc2_w,  wfc2,  (size_t)MLPDIM * CDIM);

    // 2) token conditioning: tadd = cpad @ qkvt_w^T + qkvt_b   [128, 3456]
    cpad_kernel<<<(128 * CDIM) / 256, 256, 0, stream>>>(c, cpad);
    gemm_f16_wmma<EPI_TADD><<<dim3(3 * CDIM / BN, 1), 256, 0, stream>>>(
        cpad, wqkvt, qkvt_b, nullptr, nullptr, tadd, nullptr,
        nullptr, nullptr, nullptr, 128, 3 * CDIM, CDIM);

    // 3) LN1 + zero q/k (for the DH->96 zero padding)
    ln_kernel<<<ROWS, 256, 0, stream>>>(x, hbuf);
    fill_zero_u32<<<8192, 256, 0, stream>>>((unsigned int*)qbuf,
                                            (2 * QSZ) / sizeof(unsigned int));

    // 4) qkv GEMM with fused bias + token add + q/k/vT scatter
    gemm_f16_wmma<EPI_QKV><<<dim3(3 * CDIM / BN, ROWS / BM), 256, 0, stream>>>(
        hbuf, wqkv, qkv_b, tadd, nullptr, nullptr, nullptr,
        qbuf, kbuf, vtb, ROWS, 3 * CDIM, CDIM);

    // 5) relative-position bias matrix (shared across batches)
    bias_pre_kernel<<<(HEADS * TOK_N * TOK_N) / 256, 256, 0, stream>>>(rpb, rel_idx, biasm);

    // 6) attention
    attn_kernel<<<BATCH * HEADS, 128, 0, stream>>>(qbuf, kbuf, vtb, biasm, obuf);

    // 7) proj GEMM + residual -> x1
    gemm_f16_wmma<EPI_PROJ><<<dim3(CDIM / BN, ROWS / BM), 256, 0, stream>>>(
        obuf, wproj, proj_b, nullptr, x, x1, nullptr,
        nullptr, nullptr, nullptr, ROWS, CDIM, CDIM);

    // 8) LN2 -> h2 (reuse hbuf)
    ln_kernel<<<ROWS, 256, 0, stream>>>(x1, hbuf);

    // 9) fc1 GEMM + GELU -> m1 (f16, reuses q/k/vT space)
    gemm_f16_wmma<EPI_FC1><<<dim3(MLPDIM / BN, ROWS / BM), 256, 0, stream>>>(
        hbuf, wfc1, fc1_b, nullptr, nullptr, nullptr, m1,
        nullptr, nullptr, nullptr, ROWS, MLPDIM, CDIM);

    // 10) fc2 GEMM + residual -> d_out
    gemm_f16_wmma<EPI_FC2><<<dim3(CDIM / BN, ROWS / BM), 256, 0, stream>>>(
        m1, wfc2, fc2_b, nullptr, x1, out, nullptr,
        nullptr, nullptr, nullptr, ROWS, CDIM, MLPDIM);
}
